// LocalAttention_50964081934690
// MI455X (gfx1250) — matrix operand order repaired and checked by reading, not compiled, not run on hardware
//
#include <hip/hip_runtime.h>

typedef __attribute__((ext_vector_type(16))) _Float16 v16h;
typedef __attribute__((ext_vector_type(8)))  _Float16 v8h;
typedef __attribute__((ext_vector_type(4)))  _Float16 v4h;
typedef __attribute__((ext_vector_type(8)))  float    v8f;

typedef unsigned int u32;
typedef __attribute__((ext_vector_type(4))) u32 v4u;
typedef __attribute__((ext_vector_type(8))) int v8i;
typedef __attribute__((ext_vector_type(4))) int v4i;

#define BATCH     2
#define SEQ       2048
#define HID       1024
#define NHEADS    16
#define HEAD_DIM  64
#define HALF_W    128
#define NEG_BIG   (-1.0e30f)

// LDS layout for the attention kernel (dynamic shared memory only, so these
// byte offsets ARE the ds/TDM addresses):
//   [0,      4K)  : P tiles, 1 KB per wave (16x32 f16)
//   [4K,    36K)  : K tiles, per wave 2 buffers of 4 KB (32 keys x 64 hd)
//   [36K,   68K)  : V tiles, per wave 2 buffers of 4 KB (64 hd x 32 keys)
#define LDS_PBUF_BYTES   (4 * 1024)
#define LDS_KBUF_BASE    (LDS_PBUF_BYTES)
#define LDS_VBUF_BASE    (LDS_KBUF_BASE + 8 * 4096)
#define LDS_TOTAL_BYTES  (LDS_VBUF_BASE + 8 * 4096)

__device__ __forceinline__ v16h cat8(v8h a, v8h b) {
    return __builtin_shufflevector(a, b, 0, 1, 2, 3, 4, 5, 6, 7,
                                         8, 9, 10, 11, 12, 13, 14, 15);
}

// ---------------------------------------------------------------------------
// TDM: async 2-D tile load (f16 elements) global -> LDS.
// Descriptor packing per CDNA5 ISA sec. 8 (D# groups 0/1; groups 2/3 unused
// for 2-D tiles).  Row X (dim0) contiguous in memory with stride `stride0`
// elements between rows; lands row-major and fully packed in LDS.
// ---------------------------------------------------------------------------
__device__ __forceinline__ void tdm_load_tile_f16(u32 lds_byte_off, const void* gptr,
                                                  u32 tile_d0, u32 tile_d1,
                                                  u32 tensor_d0, u32 tensor_d1,
                                                  u32 stride0)
{
    const unsigned long long ga = (unsigned long long)(uintptr_t)gptr;
    v4u g0;
    g0[0] = 1u;                                  // count=1 (valid), user mode
    g0[1] = lds_byte_off;                        // lds_addr (bytes)
    g0[2] = (u32)(ga & 0xFFFFFFFFull);           // global_addr[31:0]
    g0[3] = (u32)(ga >> 32) | (2u << 30);        // global_addr[56:32] | type=2

    v8i g1;
    g1[0] = (int)(1u << 16);                     // workgroup_mask=0, data_size=1 (2B)
    g1[1] = (int)((tensor_d0 & 0xFFFFu) << 16);                       // dim0[15:0]
    g1[2] = (int)(((tensor_d0 >> 16) & 0xFFFFu) |
                  ((tensor_d1 & 0xFFFFu) << 16));                     // dim0[31:16] | dim1[15:0]
    g1[3] = (int)(((tensor_d1 >> 16) & 0xFFFFu) | (tile_d0 << 16));   // dim1[31:16] | tile_dim0
    g1[4] = (int)(tile_d1);                                           // tile_dim1 | tile_dim2=0
    g1[5] = (int)stride0;                                             // dim0_stride[31:0]
    g1[6] = 0;                                                        // dim0_stride[47:32]
    g1[7] = 0;
    const v4i z4 = {0, 0, 0, 0};
    const v8i z8 = {0, 0, 0, 0, 0, 0, 0, 0};
    __builtin_amdgcn_tensor_load_to_lds(g0, g1, z4, z4, z8, 0);
}

// ---------------------------------------------------------------------------
// Prep A: x (f32) -> xh (f16), elementwise, 4 elements/thread.
// ---------------------------------------------------------------------------
__global__ void cvt_x_kernel(const float* __restrict__ x, _Float16* __restrict__ xh)
{
    const size_t idx = ((size_t)blockIdx.x * blockDim.x + threadIdx.x) * 4;
    const float4 v = *(const float4*)(x + idx);
    v4h o;
    o[0] = (_Float16)v.x; o[1] = (_Float16)v.y;
    o[2] = (_Float16)v.z; o[3] = (_Float16)v.w;
    *(v4h*)(xh + idx) = o;
}

// ---------------------------------------------------------------------------
// Prep B: W (f32, [k][n]) -> WT (f16, [n][k]) via LDS tile transpose.
// grid = (HID/16, HID/16, 4), block = (16,16).
// ---------------------------------------------------------------------------
__global__ void cvt_w_kernel(const float* __restrict__ Wq, const float* __restrict__ Wk,
                             const float* __restrict__ Wv, const float* __restrict__ Wo,
                             _Float16* __restrict__ WTq, _Float16* __restrict__ WTk,
                             _Float16* __restrict__ WTv, _Float16* __restrict__ WTo)
{
    __shared__ float t[16][17];
    const int mat = blockIdx.z;
    const float* W  = (mat == 0) ? Wq : (mat == 1) ? Wk : (mat == 2) ? Wv : Wo;
    _Float16*   WT  = (mat == 0) ? WTq : (mat == 1) ? WTk : (mat == 2) ? WTv : WTo;

    const int k = blockIdx.x * 16 + threadIdx.y;
    const int n = blockIdx.y * 16 + threadIdx.x;
    t[threadIdx.y][threadIdx.x] = W[(size_t)k * HID + n];
    __syncthreads();
    const int nn = blockIdx.y * 16 + threadIdx.y;
    const int kk = blockIdx.x * 16 + threadIdx.x;
    WT[(size_t)nn * HID + kk] = (_Float16)t[threadIdx.x][threadIdx.y];
}

// ---------------------------------------------------------------------------
// Kernel 1: QKV projections from f16 operands.  Each wave computes a 16x64
// strip (4 C tiles) reusing one A fragment per k-step.  Q scaled by 1/8.
// V stored head-transposed: Vt[((b*16+h)*64+hd)*SEQ + s].
// grid = (M/16, HID/256, 3), block = 128 (4 waves).
// ---------------------------------------------------------------------------
__global__ void qkv_proj_kernel(const _Float16* __restrict__ xh,
                                const _Float16* __restrict__ WTq,
                                const _Float16* __restrict__ WTk,
                                const _Float16* __restrict__ WTv,
                                const float* __restrict__ bq,
                                const float* __restrict__ bk,
                                const float* __restrict__ bv,
                                _Float16* __restrict__ Qh,
                                _Float16* __restrict__ Kh,
                                _Float16* __restrict__ Vt)
{
    const int lane = threadIdx.x & 31;
    const int wave = threadIdx.x >> 5;
    const int lo   = lane & 15;
    const int hi   = lane >> 4;
    const int rowTile = blockIdx.x;
    const int colBase = (blockIdx.y * 4 + wave) * 64;
    const int mat     = blockIdx.z;

    const _Float16* WT = (mat == 0) ? WTq : (mat == 1) ? WTk : WTv;
    const float*    bs = (mat == 0) ? bq : (mat == 1) ? bk : bv;

    const int r    = rowTile * 16 + lo;
    const int koff = hi * 8;

    v8f c[4] = {};
    for (int k0 = 0; k0 < HID; k0 += 32) {
        const _Float16* ar = xh + (size_t)r * HID + k0;
        const v16h a = *(const v16h*)(ar + hi * 16);
        const int kb = k0 + hi * 16;
#pragma unroll
        for (int nt = 0; nt < 4; ++nt) {
            const int n = colBase + nt * 16 + lo;
            const v16h b = *(const v16h*)(WT + (size_t)n * HID + kb);
            c[nt] = __builtin_amdgcn_wmma_f32_16x16x32_f16(false, a, false, b,
                                                           (short)0, c[nt], false, false);
        }
    }

    const float sc = (mat == 0) ? 0.125f : 1.0f;
#pragma unroll
    for (int nt = 0; nt < 4; ++nt) {
        const int n = colBase + nt * 16 + lo;
        const float bias = bs[n];
        if (mat == 2) {
            const int head = n >> 6, hd = n & 63;
#pragma unroll
            for (int i = 0; i < 8; ++i) {
                const int row = rowTile * 16 + i + 8 * hi;
                const int bb = row >> 11, s = row & (SEQ - 1);
                Vt[((size_t)((bb * NHEADS + head) * HEAD_DIM + hd)) * SEQ + s]
                    = (_Float16)(c[nt][i] + bias);
            }
        } else {
            _Float16* Y = (mat == 0) ? Qh : Kh;
#pragma unroll
            for (int i = 0; i < 8; ++i) {
                const int m = i + 8 * hi;
                Y[(size_t)(rowTile * 16 + m) * HID + n] = (_Float16)((c[nt][i] + bias) * sc);
            }
        }
    }
}

// ---------------------------------------------------------------------------
// Kernel 2: banded flash attention with TDM-staged K/V tiles.
// One wave per (batch, head, 16-query tile); 4 heads per block (identical
// trip counts).  Per 32-key chunk the wave's K (32x64) and V (64x32) tiles
// are DMA'd into LDS by the Tensor Data Mover, double-buffered so chunk j+32
// streams in while chunk j computes.  grid = (SEQ/16, NHEADS/4, BATCH).
// ---------------------------------------------------------------------------
__global__ void local_attn_kernel(const _Float16* __restrict__ Qh,
                                  const _Float16* __restrict__ Kh,
                                  const _Float16* __restrict__ Vt,
                                  _Float16* __restrict__ Ah)
{
    extern __shared__ _Float16 smem[];   // dynamic only -> offsets start at 0

    const int lane = threadIdx.x & 31;
    const int wave = threadIdx.x >> 5;
    const int lo   = lane & 15;
    const int hi   = lane >> 4;
    const int qt   = blockIdx.x;
    const int head = blockIdx.y * 4 + wave;
    const int bb   = blockIdx.z;
    const int koff = hi * 8;

    _Float16* pbuf = smem + (wave * 1024) / 2;   // 16x32 f16 per wave

    // Q fragments over hd 0..31 and 32..63
    v16h aq0, aq1;
    {
        const int qrow = qt * 16 + lo;
        const _Float16* Qr = Qh + ((size_t)(bb * SEQ + qrow)) * HID + head * HEAD_DIM;
        aq0 = *(const v16h*)(Qr + hi * 16);
        aq1 = *(const v16h*)(Qr + 32 + hi * 16);
    }

    v8f oacc[4] = {};
    float rm[8], rl[8];
#pragma unroll
    for (int i = 0; i < 8; ++i) { rm[i] = NEG_BIG; rl[i] = 0.0f; }

    int wstart = qt * 16 - HALF_W; if (wstart < 0) wstart = 0;
    const int s0 = wstart & ~31;
    int send = qt * 16 + 15 + HALF_W; if (send > SEQ - 1) send = SEQ - 1;

    const _Float16* Kbase = Kh + (size_t)(bb * SEQ) * HID + head * HEAD_DIM;
    const _Float16* Vbase = Vt + (size_t)(bb * NHEADS + head) * HEAD_DIM * SEQ;

    // issue TDM for chunk 0 (buffer 0): K 32x64 (stride HID), V 64x32 (stride SEQ)
    tdm_load_tile_f16(LDS_KBUF_BASE + (wave * 2 + 0) * 4096,
                      Kbase + (size_t)s0 * HID, 64, 32, 64, 32, HID);
    tdm_load_tile_f16(LDS_VBUF_BASE + (wave * 2 + 0) * 4096,
                      Vbase + s0, 32, 64, 32, 64, SEQ);

    int buf = 0;
    for (int j = s0; j <= send; j += 32, buf ^= 1) {
        // prefetch next chunk into the other buffer, then wait for current
        if (j + 32 <= send) {
            tdm_load_tile_f16(LDS_KBUF_BASE + (wave * 2 + (buf ^ 1)) * 4096,
                              Kbase + (size_t)(j + 32) * HID, 64, 32, 64, 32, HID);
            tdm_load_tile_f16(LDS_VBUF_BASE + (wave * 2 + (buf ^ 1)) * 4096,
                              Vbase + (j + 32), 32, 64, 32, 64, SEQ);
            __builtin_amdgcn_s_wait_tensorcnt(2);   // in-order: current pair done
        } else {
            __builtin_amdgcn_s_wait_tensorcnt(0);
        }
        asm volatile("" ::: "memory");              // keep ds reads below the wait

        const _Float16* kbuf = smem + (LDS_KBUF_BASE + (wave * 2 + buf) * 4096) / 2;
        const _Float16* vbuf = smem + (LDS_VBUF_BASE + (wave * 2 + buf) * 4096) / 2;

        // ---- scores: two 16x16 C tiles over the 32-key chunk (K from LDS) ----
        const v16h kb00 = *(const v16h*)(kbuf + (size_t)lo * 64 + hi * 16);
        const v16h kb01 = *(const v16h*)(kbuf + (size_t)lo * 64 + hi * 16 + 32);
        const v16h kb10 = *(const v16h*)(kbuf + (size_t)(16 + lo) * 64 + hi * 16);
        const v16h kb11 = *(const v16h*)(kbuf + (size_t)(16 + lo) * 64 + hi * 16 + 32);

        v8f cs0 = {}, cs1 = {};
        cs0 = __builtin_amdgcn_wmma_f32_16x16x32_f16(false, aq0, false, kb00, (short)0, cs0, false, false);
        cs0 = __builtin_amdgcn_wmma_f32_16x16x32_f16(false, aq1, false, kb01, (short)0, cs0, false, false);
        cs1 = __builtin_amdgcn_wmma_f32_16x16x32_f16(false, aq0, false, kb10, (short)0, cs1, false, false);
        cs1 = __builtin_amdgcn_wmma_f32_16x16x32_f16(false, aq1, false, kb11, (short)0, cs1, false, false);

        // ---- band mask + online softmax ----
#pragma unroll
        for (int i = 0; i < 8; ++i) {
            const int m   = i + 8 * hi;
            const int q   = qt * 16 + m;
            const int k0i = j + lo;
            const int k1i = j + 16 + lo;
            int d0 = q - k0i; d0 = d0 < 0 ? -d0 : d0;
            int d1 = q - k1i; d1 = d1 < 0 ? -d1 : d1;
            const float sv0 = (d0 <= HALF_W) ? cs0[i] : NEG_BIG;
            const float sv1 = (d1 <= HALF_W) ? cs1[i] : NEG_BIG;

            float mx = fmaxf(sv0, sv1);
#pragma unroll
            for (int d = 1; d < 16; d <<= 1) mx = fmaxf(mx, __shfl_xor(mx, d, 32));
            const float mnew = fmaxf(rm[i], mx);
            const float scal = __expf(rm[i] - mnew);
            const float p0   = __expf(sv0 - mnew);
            const float p1   = __expf(sv1 - mnew);
            float sum = p0 + p1;
#pragma unroll
            for (int d = 1; d < 16; d <<= 1) sum += __shfl_xor(sum, d, 32);
            rl[i] = rl[i] * scal + sum;
            rm[i] = mnew;
#pragma unroll
            for (int nt = 0; nt < 4; ++nt) oacc[nt][i] *= scal;

            pbuf[m * 32 + lo]      = (_Float16)p0;
            pbuf[m * 32 + 16 + lo] = (_Float16)p1;
        }
        __syncthreads();

        // re-layout P: C-fragment -> A-fragment (aligned ds reads)
        const v16h ap = *(const v16h*)(pbuf + lo * 32 + hi * 16);
        __syncthreads();

        // ---- O += P (16x32) . V (32x64); V fragments from LDS ----
#pragma unroll
        for (int nt = 0; nt < 4; ++nt) {
            const v16h vb = *(const v16h*)(vbuf + (size_t)(nt * 16 + lo) * 32 + hi * 16);
            oacc[nt] = __builtin_amdgcn_wmma_f32_16x16x32_f16(false, ap, false, vb,
                                                              (short)0, oacc[nt], false, false);
        }
    }

    // normalize and store attention output (f16)
#pragma unroll
    for (int nt = 0; nt < 4; ++nt) {
#pragma unroll
        for (int i = 0; i < 8; ++i) {
            const int m = i + 8 * hi;
            const float val = oacc[nt][i] / rl[i];
            Ah[((size_t)(bb * SEQ + qt * 16 + m)) * HID + head * HEAD_DIM + nt * 16 + lo]
                = (_Float16)val;
        }
    }
}

// ---------------------------------------------------------------------------
// Kernel 3: output projection.  out = Ah @ Wo + bo (f32 result).
// grid = (M/16, HID/256), block = 128.
// ---------------------------------------------------------------------------
__global__ void out_proj_kernel(const _Float16* __restrict__ Ah,
                                const _Float16* __restrict__ WTo,
                                const float* __restrict__ bo,
                                float* __restrict__ out)
{
    const int lane = threadIdx.x & 31;
    const int wave = threadIdx.x >> 5;
    const int lo   = lane & 15;
    const int hi   = lane >> 4;
    const int rowTile = blockIdx.x;
    const int colBase = (blockIdx.y * 4 + wave) * 64;

    const int r    = rowTile * 16 + lo;
    const int koff = hi * 8;

    v8f c[4] = {};
    for (int k0 = 0; k0 < HID; k0 += 32) {
        const _Float16* ar = Ah + (size_t)r * HID + k0;
        const v16h a = *(const v16h*)(ar + hi * 16);
        const int kb = k0 + hi * 16;
#pragma unroll
        for (int nt = 0; nt < 4; ++nt) {
            const int n = colBase + nt * 16 + lo;
            const v16h b = *(const v16h*)(WTo + (size_t)n * HID + kb);
            c[nt] = __builtin_amdgcn_wmma_f32_16x16x32_f16(false, a, false, b,
                                                           (short)0, c[nt], false, false);
        }
    }

#pragma unroll
    for (int nt = 0; nt < 4; ++nt) {
        const int n = colBase + nt * 16 + lo;
        const float bias = bo[n];
#pragma unroll
        for (int i = 0; i < 8; ++i) {
            const int m = i + 8 * hi;
            out[(size_t)(rowTile * 16 + m) * HID + n] = c[nt][i] + bias;
        }
    }
}

// ---------------------------------------------------------------------------
extern "C" void kernel_launch(void* const* d_in, const int* in_sizes, int n_in,
                              void* d_out, int out_size, void* d_ws, size_t ws_size,
                              hipStream_t stream)
{
    (void)in_sizes; (void)n_in; (void)out_size; (void)ws_size;

    const float* x  = (const float*)d_in[0];
    const float* Wq = (const float*)d_in[1];
    const float* bq = (const float*)d_in[2];
    const float* Wk = (const float*)d_in[3];
    const float* bk = (const float*)d_in[4];
    const float* Wv = (const float*)d_in[5];
    const float* bv = (const float*)d_in[6];
    const float* Wo = (const float*)d_in[7];
    const float* bo = (const float*)d_in[8];
    float* out = (float*)d_out;

    const size_t MH = (size_t)BATCH * SEQ * HID;
    const size_t WW = (size_t)HID * HID;
    _Float16* xh  = (_Float16*)d_ws;
    _Float16* Qh  = xh + MH;
    _Float16* Kh  = Qh + MH;
    _Float16* Vt  = Kh + MH;
    _Float16* Ah  = Vt + MH;
    _Float16* WTq = Ah + MH;
    _Float16* WTk = WTq + WW;
    _Float16* WTv = WTk + WW;
    _Float16* WTo = WTv + WW;

    cvt_x_kernel<<<dim3(MH / (256 * 4)), 256, 0, stream>>>(x, xh);
    cvt_w_kernel<<<dim3(HID / 16, HID / 16, 4), dim3(16, 16), 0, stream>>>(
        Wq, Wk, Wv, Wo, WTq, WTk, WTv, WTo);

    dim3 g1(BATCH * SEQ / 16, HID / 256, 3);
    qkv_proj_kernel<<<g1, 128, 0, stream>>>(xh, WTq, WTk, WTv, bq, bk, bv, Qh, Kh, Vt);

    dim3 g2(SEQ / 16, NHEADS / 4, BATCH);
    local_attn_kernel<<<g2, 128, LDS_TOTAL_BYTES, stream>>>(Qh, Kh, Vt, Ah);

    dim3 g3(BATCH * SEQ / 16, HID / 256);
    out_proj_kernel<<<g3, 128, LDS_TOTAL_BYTES, stream>>>(Ah, WTo, bo, out);
}
